// PseudoTokenGridEncoder_78932908966060
// MI455X (gfx1250) — matrix operand order repaired and checked by reading, not compiled, not run on hardware
//
#include <hip/hip_runtime.h>

// ---------------- problem constants (from reference setup_inputs) -----------
#define BB   4
#define UU   4096
#define EE   128
#define GG   64
#define SS   (GG * GG)          // 4096 cells per batch
#define NC   (BB * SS)          // 16384 cells total
#define NT   (BB * UU)          // 16384 off-grid tokens total
#define HH   8
#define DH   16                 // head dim, scale = 1/sqrt(16) = 0.25

// ---------------- bf16 helpers / WMMA types --------------------------------
typedef __attribute__((ext_vector_type(16))) __bf16 v16bf;
typedef __attribute__((ext_vector_type(8)))  __bf16 v8bf;
typedef __attribute__((ext_vector_type(8)))  float  v8f;

__device__ __forceinline__ __bf16 f2bf(float f) {
  unsigned int u = __float_as_uint(f);
  unsigned short h = (unsigned short)((u + 0x7FFFu + ((u >> 16) & 1u)) >> 16);
  return __builtin_bit_cast(__bf16, h);
}

// ---------------- kernel 0: zero int buffer --------------------------------
__global__ void zero_i32_kernel(int* __restrict__ p, int n) {
  int i = blockIdx.x * 256 + threadIdx.x;
  if (i < n) p[i] = 0;
}

// ---------------- kernel 1: nearest cell (L1 argmin) + counts --------------
__global__ __launch_bounds__(256) void nearest_kernel(
    const float* __restrict__ xc_off,   // [B,U,2]
    const float* __restrict__ xc_on,    // [B,S,2]
    int* __restrict__ nearest,          // [B,U]
    int* __restrict__ counts) {         // [B,S]
  __shared__ float xs[SS];
  __shared__ float ys[SS];
  int b     = blockIdx.x >> 4;          // 16 chunks of 256 per batch
  int chunk = blockIdx.x & 15;
  const float* on = xc_on + (size_t)b * SS * 2;
  for (int i = threadIdx.x; i < SS; i += 256) {
    xs[i] = on[2 * i + 0];
    ys[i] = on[2 * i + 1];
  }
  __syncthreads();
  int u = chunk * 256 + threadIdx.x;
  const float* po = xc_off + ((size_t)b * UU + u) * 2;
  float px = po[0], py = po[1];
  float best = 3.0e38f;
  int bi = 0;
  for (int s = 0; s < SS; ++s) {
    float d = fabsf(px - xs[s]) + fabsf(py - ys[s]);
    if (d < best) { best = d; bi = s; }   // strict '<' == argmin first-index
  }
  nearest[b * UU + u] = bi;
  atomicAdd(&counts[b * SS + bi], 1);     // integer atomic: deterministic
}

// ---------------- kernel 2: exclusive scan over NC counts (1 block) --------
__global__ __launch_bounds__(256) void scan_kernel(
    const int* __restrict__ counts, int* __restrict__ offsets) {
  __shared__ int part[256];
  int tid = threadIdx.x;
  int base = tid * (NC / 256);            // 64 per thread
  int s = 0;
  for (int i = 0; i < NC / 256; ++i) s += counts[base + i];
  part[tid] = s;
  __syncthreads();
  if (tid == 0) {
    int run = 0;
    for (int i = 0; i < 256; ++i) { int t = part[i]; part[i] = run; run += t; }
  }
  __syncthreads();
  int run = part[tid];
  for (int i = 0; i < NC / 256; ++i) { offsets[base + i] = run; run += counts[base + i]; }
}

// ---------------- kernel 3: deterministic rank + scatter to CSR lists ------
__global__ __launch_bounds__(256) void scatter_kernel(
    const int* __restrict__ nearest,    // [B,U]
    const int* __restrict__ offsets,    // [B*S]
    int* __restrict__ tok_list) {       // [NT]
  __shared__ int idxbuf[UU];
  int b     = blockIdx.x >> 4;
  int chunk = blockIdx.x & 15;
  for (int i = threadIdx.x; i < UU; i += 256) idxbuf[i] = nearest[b * UU + i];
  __syncthreads();
  int u  = chunk * 256 + threadIdx.x;
  int my = idxbuf[u];
  int rank = 0;                          // == reference cumcount
  for (int v = 0; v < u; ++v) rank += (idxbuf[v] == my) ? 1 : 0;
  tok_list[offsets[b * SS + my] + rank] = u;
}

// ---------------- kernel 4: build "last" token (on-grid or fake) -----------
__global__ void build_last_kernel(
    const float* __restrict__ zc_on,    // [B*S,E]
    const float* __restrict__ fake,     // [E]
    const int* __restrict__ flag,       // scalar ignore_on_grid
    float* __restrict__ last) {         // [B*S,E]
  int i = blockIdx.x * 256 + threadIdx.x;   // NC*EE total
  int ig = flag[0];
  last[i] = ig ? fake[i & (EE - 1)] : zc_on[i];
}

// ---------------- kernel 5: C[M,128] = A[M,128] @ W[128,128] via WMMA ------
// block = 256 threads (8 waves); each wave -> one 16x16 tile; block -> 16x128.
__global__ __launch_bounds__(256) void gemm128_wmma_kernel(
    const float* __restrict__ A, const float* __restrict__ W,
    float* __restrict__ C) {
  __shared__ __bf16 Wt[EE][EE + 8];     // W transposed: Wt[n][k], 16B-aligned rows
  __shared__ __bf16 At[16][EE + 8];     // A tile (16 rows)
  int tid = threadIdx.x;
  int m0  = blockIdx.x * 16;

  for (int i = tid; i < EE * EE; i += 256) {
    int k = i >> 7, n = i & (EE - 1);
    Wt[n][k] = f2bf(W[i]);
  }
  for (int i = tid; i < 16 * EE; i += 256) {
    int r = i >> 7, k = i & (EE - 1);
    At[r][k] = f2bf(A[(size_t)(m0 + r) * EE + k]);
  }
  __syncthreads();

  int lane = tid & 31;
  int wid  = tid >> 5;                  // 0..7
  int n0   = wid * 16;
  int row  = lane & 15;                 // A: M index / B: N index
  int half = lane >> 4;                 // lane group 0/1

  v8f c = {};
#pragma unroll
  for (int kb = 0; kb < EE; kb += 32) {
    v16bf a, bm;
    // A 16x32 bf16 layout: lanes<16 K{0..7,16..23}, lanes>=16 K{8..15,24..31}
    ((v8bf*)&a)[0]  = *(const v8bf*)&At[row][kb + half * 8];
    ((v8bf*)&a)[1]  = *(const v8bf*)&At[row][kb + 16 + half * 8];
    // B 32x16 bf16 layout: lanes<16 K{0..7,16..23}, lanes>=16 K{8..15,24..31}, N=lane%16
    ((v8bf*)&bm)[0] = *(const v8bf*)&Wt[n0 + row][kb + half * 8];
    ((v8bf*)&bm)[1] = *(const v8bf*)&Wt[n0 + row][kb + 16 + half * 8];
    c = __builtin_amdgcn_wmma_f32_16x16x32_bf16(
            false, a, false, bm, (short)0, c, false, false);
  }
  // C/D: VGPR r -> M = r + 8*half, N = lane%16
  int col = n0 + row;
#pragma unroll
  for (int r = 0; r < 8; ++r)
    C[(size_t)(m0 + half * 8 + r) * EE + col] = c[r];
}

// ---------------- kernel 6: per-(cell,head) online-softmax attention -------
__global__ __launch_bounds__(256) void attn_kernel(
    const float* __restrict__ Q,        // [S,E]  (latents projected)
    const float* __restrict__ Koff, const float* __restrict__ Voff, // [NT,E]
    const float* __restrict__ Kon,  const float* __restrict__ Von,  // [NC,E]
    const int* __restrict__ offsets, const int* __restrict__ counts,
    const int* __restrict__ tok_list,
    float* __restrict__ out) {          // [NC,E]
  int g    = blockIdx.x * 256 + threadIdx.x;  // NC*HH threads
  int cell = g >> 3;
  int h    = g & 7;
  int b    = cell >> 12;                // / SS
  int s    = cell & (SS - 1);

  float qv[DH];
  const float* qp = Q + (size_t)s * EE + h * DH;
#pragma unroll
  for (int j = 0; j < DH; ++j) qv[j] = qp[j];

  const float scale = 0.25f;            // 1/sqrt(16)
  // last slot (on-grid / fake) is always valid -> seed the online softmax
  const float* kp = Kon + (size_t)cell * EE + h * DH;
  const float* vp = Von + (size_t)cell * EE + h * DH;
  float sc = 0.f;
#pragma unroll
  for (int j = 0; j < DH; ++j) sc += qv[j] * kp[j];
  float m = sc * scale;
  float den = 1.0f;
  float num[DH];
#pragma unroll
  for (int j = 0; j < DH; ++j) num[j] = vp[j];

  int off = offsets[cell], cnt = counts[cell];
  for (int t = 0; t < cnt; ++t) {
    int u = tok_list[off + t];
    const float* kk = Koff + ((size_t)b * UU + u) * EE + h * DH;
    const float* vv = Voff + ((size_t)b * UU + u) * EE + h * DH;
    float d = 0.f;
#pragma unroll
    for (int j = 0; j < DH; ++j) d += qv[j] * kk[j];
    d *= scale;
    if (d <= m) {
      float w = __expf(d - m);
      den += w;
#pragma unroll
      for (int j = 0; j < DH; ++j) num[j] += w * vv[j];
    } else {
      float f = __expf(m - d);
      den = den * f + 1.0f;
#pragma unroll
      for (int j = 0; j < DH; ++j) num[j] = num[j] * f + vv[j];
      m = d;
    }
  }
  float inv = 1.0f / den;
  float* op = out + (size_t)cell * EE + h * DH;
#pragma unroll
  for (int j = 0; j < DH; ++j) op[j] = num[j] * inv;
}

// ---------------- host launcher --------------------------------------------
extern "C" void kernel_launch(void* const* d_in, const int* in_sizes, int n_in,
                              void* d_out, int out_size, void* d_ws, size_t ws_size,
                              hipStream_t stream) {
  const float* xc_off  = (const float*)d_in[0];   // [B,U,2]
  const float* xc_on   = (const float*)d_in[1];   // [B,G,G,2]
  const float* zc_off  = (const float*)d_in[2];   // [B,U,E]
  const float* zc_on   = (const float*)d_in[3];   // [B,G,G,E]
  const float* latents = (const float*)d_in[4];   // [S,E]
  const float* fake    = (const float*)d_in[5];   // [E]
  const float* Wq      = (const float*)d_in[6];
  const float* Wk      = (const float*)d_in[7];
  const float* Wv      = (const float*)d_in[8];
  const float* Wo      = (const float*)d_in[9];
  const int*   ignore  = (const int*)d_in[10];
  float* out = (float*)d_out;                     // [B,G,G,E] = [NC,E]

  char* ws = (char*)d_ws;
  int*   nearest  = (int*)ws;                ws += (size_t)NT * 4;
  int*   counts   = (int*)ws;                ws += (size_t)NC * 4;
  int*   offsets  = (int*)ws;                ws += (size_t)NC * 4;
  int*   tok_list = (int*)ws;                ws += (size_t)NT * 4;
  float* Qb       = (float*)ws;              ws += (size_t)SS * EE * 4;
  float* Koff     = (float*)ws;              ws += (size_t)NT * EE * 4;
  float* Voff     = (float*)ws;              ws += (size_t)NT * EE * 4;
  float* Kon      = (float*)ws;              ws += (size_t)NC * EE * 4;
  float* Von      = (float*)ws;              ws += (size_t)NC * EE * 4;
  float* last     = (float*)ws;              ws += (size_t)NC * EE * 4;
  float* attnout  = last;                    // reuse: 'last' dead after K_on/V_on

  // 1) counts = 0
  zero_i32_kernel<<<NC / 256, 256, 0, stream>>>(counts, NC);
  // 2) nearest + counts
  nearest_kernel<<<BB * (UU / 256), 256, 0, stream>>>(xc_off, xc_on, nearest, counts);
  // 3) exclusive scan
  scan_kernel<<<1, 256, 0, stream>>>(counts, offsets);
  // 4) deterministic scatter (rank == reference cumcount)
  scatter_kernel<<<BB * (UU / 256), 256, 0, stream>>>(nearest, offsets, tok_list);
  // 5) last token embedding
  build_last_kernel<<<(NC * EE) / 256, 256, 0, stream>>>(zc_on, fake, ignore, last);
  // 6) projections (WMMA bf16 GEMMs)
  gemm128_wmma_kernel<<<SS / 16, 256, 0, stream>>>(latents, Wq, Qb);
  gemm128_wmma_kernel<<<NT / 16, 256, 0, stream>>>(zc_off, Wk, Koff);
  gemm128_wmma_kernel<<<NT / 16, 256, 0, stream>>>(zc_off, Wv, Voff);
  gemm128_wmma_kernel<<<NC / 16, 256, 0, stream>>>(last, Wk, Kon);
  gemm128_wmma_kernel<<<NC / 16, 256, 0, stream>>>(last, Wv, Von);
  // 7) masked cross-attention (online softmax per cell/head)
  attn_kernel<<<(NC * HH) / 256, 256, 0, stream>>>(Qb, Koff, Voff, Kon, Von,
                                                   offsets, counts, tok_list, attnout);
  // 8) output projection directly into d_out
  gemm128_wmma_kernel<<<NC / 16, 256, 0, stream>>>(attnout, Wo, out);
}
